// WaveNet_36155034698369
// MI455X (gfx1250) — matrix operand order repaired and checked by reading, not compiled, not run on hardware
//
#include <hip/hip_runtime.h>
#include <hip/hip_bf16.h>

// ---------------- model constants (match reference) ----------------
#define NUM_LAYERS 30
#define Bsz 32
#define Lsz 4096
#define RCc 64     // residual channels
#define DCc 64     // dilated channels
#define SCc 64     // skip channels
#define ECc 128    // end channels
#define INC 32     // input channels

#define TILE_N 128 // positions per workgroup
#define NWAVES 8   // 256 threads = 8 wave32

typedef __bf16 bf16_t;
typedef __attribute__((ext_vector_type(16))) __bf16 v16bf;
typedef __attribute__((ext_vector_type(8)))  __bf16 v8bf;
typedef __attribute__((ext_vector_type(8)))  float  v8f;
typedef __attribute__((ext_vector_type(4)))  float  v4f;
typedef unsigned int u32x4 __attribute__((ext_vector_type(4)));
typedef int          i32x4 __attribute__((ext_vector_type(4)));
typedef int          i32x8 __attribute__((ext_vector_type(8)));

__device__ __forceinline__ v8f v8f_zero() {
    v8f z;
#pragma unroll
    for (int j = 0; j < 8; ++j) z[j] = 0.0f;
    return z;
}

// WMMA D = A*B + C, bf16 inputs, f32 accumulate (v_wmma_f32_16x16x32_bf16)
__device__ __forceinline__ v8f wmma_bf16(v16bf a, v16bf b, v8f c) {
    return __builtin_amdgcn_wmma_f32_16x16x32_bf16(
        /*neg_a=*/false, a, /*neg_b=*/false, b,
        /*c_mod=*/(short)0, c, /*reuse_a=*/false, /*reuse_b=*/false);
}

// Tensor Data Mover: flat 1-D byte copy global -> LDS (D# count=1, data_size=1B,
// tile_dim0 = tensor_dim0 = nbytes). Tracked with TENSORcnt.
__device__ __forceinline__ void tdm_copy_to_lds(const void* gptr, void* lptr, int nbytes) {
    unsigned long long ga = (unsigned long long)gptr;
    unsigned int la = (unsigned int)(unsigned long long)lptr; // low 32b = LDS offset
    u32x4 g0;
    g0[0] = 1u;                                        // count=1 (valid user D#)
    g0[1] = la;                                        // lds_addr
    g0[2] = (unsigned int)(ga & 0xFFFFFFFFu);          // global_addr[31:0]
    g0[3] = (unsigned int)((ga >> 32) & 0x01FFFFFFu)   // global_addr[56:32]
          | (2u << 30);                                // type = 2 ("image")
    i32x8 g1;
    g1[0] = 0;                                         // wg_mask=0, data_size=1B
    g1[1] = (int)((unsigned)(nbytes & 0xFFFF) << 16);  // tensor_dim0[15:0] @ [31:16]
    g1[2] = (int)((unsigned)nbytes >> 16);             // tensor_dim0[31:16]
    g1[3] = (int)((unsigned)(nbytes & 0xFFFF) << 16);  // tile_dim0 @ [31:16]
    g1[4] = 0;                                         // tile_dim1/2 unused
    g1[5] = nbytes;                                    // tensor_dim0_stride lo
    g1[6] = 0;
    g1[7] = 0;
    i32x4 gz = {0, 0, 0, 0};
#if defined(__clang_major__) && (__clang_major__ >= 23)
    i32x8 gz8 = {0, 0, 0, 0, 0, 0, 0, 0};
    __builtin_amdgcn_tensor_load_to_lds(g0, g1, gz, gz, gz8, 0);
#else
    __builtin_amdgcn_tensor_load_to_lds(g0, g1, gz, gz, 0);
#endif
}

// A-fragment (16x32 bf16) from row-major bf16 matrix (LDS or global), row stride rs.
// Layout: lane m=ln&15, half hf=ln>>4; elems 0..15 -> K=k0+hf*16+j
//             (contiguous, matches B-frag).
__device__ __forceinline__ v16bf load_afrag(const bf16_t* w, int m0, int k0, int rs) {
    const int ln = threadIdx.x & 31;
    const bf16_t* p = w + (size_t)(m0 + (ln & 15)) * rs + k0 + ((ln >> 4) << 4);
    v8bf lo = *(const v8bf*)p;
    v8bf hi = *(const v8bf*)(p + 8);
    return __builtin_shufflevector(lo, hi, 0,1,2,3,4,5,6,7,8,9,10,11,12,13,14,15);
}

// B-fragment (32x16 bf16) from bf16 activation mirror stored [t][64ch].
// Contiguous k order (same as A-frag): lane n=ln&15; lanes 0-15 hold K=0..15, lanes 16-31 hold K=16..31.
__device__ __forceinline__ v16bf load_bfrag(const bf16_t* __restrict__ hb,
                                            int t0, int ch0, int doff) {
    const int ln = threadIdx.x & 31;
    const int t  = t0 + (ln & 15) + doff;
    if (t < 0) {
        v16bf r;
#pragma unroll
        for (int j = 0; j < 16; ++j) r[j] = (bf16_t)0.0f;
        return r;
    }
    const bf16_t* p = hb + (size_t)t * RCc + ch0 + ((ln >> 4) << 4);
    v8bf lo = *(const v8bf*)p;
    v8bf hi = *(const v8bf*)(p + 8);
    return __builtin_shufflevector(lo, hi, 0,1,2,3,4,5,6,7,8,9,10,11,12,13,14,15);
}

// tanh(f) * sigmoid(g) with two v_exp_f32: (e^{2f}-1) / ((e^{2f}+1)(1+e^{-g}))
__device__ __forceinline__ float gated_act(float f, float g) {
    f = fminf(fmaxf(f, -15.0f), 15.0f);
    g = fminf(fmaxf(g, -30.0f), 30.0f);
    float tf = __expf(2.0f * f);
    float eg = __expf(-g);
    return (tf - 1.0f) / ((tf + 1.0f) * (1.0f + eg));
}

// ---------------- one WaveNet residual layer ----------------
// h_out[t] = h_in[t] + Wr * z(t) + br,   z = tanh(f) * sigmoid(g)
// f = Wf * [h(t-d); h(t)] + bf (K=128),  g likewise.
// Extra block (blockIdx.x == L/TILE_N) computes this layer's skip
// contribution at output position L (tap0-only at h[L-d]).
__global__ __launch_bounds__(256) void wavenet_layer_kernel(
    const float*  __restrict__ h_in,      // [B][L][64] f32 master
    const bf16_t* __restrict__ hb_in,     // [B][L][64] bf16 mirror
    float*        __restrict__ h_out,
    bf16_t*       __restrict__ hb_out,
    const bf16_t* __restrict__ wf,        // [64][128] bf16
    const bf16_t* __restrict__ wg,        // [64][128] bf16
    const bf16_t* __restrict__ wr,        // [64][64]  bf16
    const float* __restrict__ bf_bias,    // [64]
    const float* __restrict__ bg_bias,    // [64]
    const float* __restrict__ br_bias,    // [64]
    const float* __restrict__ dil_w_f32,  // this layer [64][64][2]
    const float* __restrict__ gate_w_f32, // this layer [64][64][2]
    const float* __restrict__ skip_w_f32, // this layer [64][64]
    const float* __restrict__ skip_bias,  // [64]
    float* __restrict__ skip_acc,         // [B][64]
    int dil)
{
    const int b = blockIdx.y;
    const int nTiles = Lsz / TILE_N;

    if ((int)blockIdx.x == nTiles) {
        // ---- skip tap at output position L: z uses only h_in[L-dil] ----
        __shared__ float zsh[DCc];
        const int o = threadIdx.x;
        const float* hv = h_in + ((size_t)b * Lsz + (Lsz - dil)) * RCc;
        if (o < DCc) {
            float af = bf_bias[o], ag = bg_bias[o];
            for (int c = 0; c < RCc; ++c) {
                float h = hv[c];
                af += dil_w_f32[(o * RCc + c) * 2] * h;
                ag += gate_w_f32[(o * RCc + c) * 2] * h;
            }
            zsh[o] = gated_act(af, ag);
        }
        __syncthreads();
        if (o < SCc) {
            float s = skip_bias[o];
            for (int c = 0; c < DCc; ++c) s += skip_w_f32[o * DCc + c] * zsh[c];
            skip_acc[b * SCc + o] += s;   // layers serialized on the stream
        }
        return;
    }

    // ---- stage this layer's weights into LDS once per block via TDM ----
    __shared__ bf16_t lds_wf[64 * 128];
    __shared__ bf16_t lds_wg[64 * 128];
    __shared__ bf16_t lds_wr[64 * 64];
    if (threadIdx.x < 32) {                 // one wave issues the 3 DMA descriptors
        tdm_copy_to_lds(wf, lds_wf, 64 * 128 * (int)sizeof(bf16_t));
        tdm_copy_to_lds(wg, lds_wg, 64 * 128 * (int)sizeof(bf16_t));
        tdm_copy_to_lds(wr, lds_wr, 64 * 64 * (int)sizeof(bf16_t));
        __builtin_amdgcn_s_wait_tensorcnt(0);
    }
    __syncthreads();

    // ---- main trunk tile: TILE_N positions, 8 waves, 1 N-tile per wave ----
    const int wave = threadIdx.x >> 5;
    const int ln   = threadIdx.x & 31;
    const int lrow = ln & 15;
    const int hf   = ln >> 4;
    const int t0   = blockIdx.x * TILE_N + wave * 16;
    const bf16_t* hbb = hb_in + (size_t)b * Lsz * RCc;
    const float*  hb  = h_in  + (size_t)b * Lsz * RCc;
    float*        ho  = h_out + (size_t)b * Lsz * RCc;
    bf16_t*       hbo = hb_out + (size_t)b * Lsz * RCc;

    v8f accf[4], accg[4];
#pragma unroll
    for (int mt = 0; mt < 4; ++mt) { accf[mt] = v8f_zero(); accg[mt] = v8f_zero(); }

    // gated convs: K = 128 (tap0 | tap1), 4 chunks of 32; B-frag shared f/g
#pragma unroll
    for (int kc = 0; kc < 4; ++kc) {
        const int k0   = kc * 32;
        const int ch0  = k0 & 63;
        const int doff = (k0 < 64) ? -dil : 0;
        v16bf bfr = load_bfrag(hbb, t0, ch0, doff);
#pragma unroll
        for (int mt = 0; mt < 4; ++mt) {
            v16bf af = load_afrag(lds_wf, mt * 16, k0, 128);
            accf[mt] = wmma_bf16(af, bfr, accf[mt]);
            v16bf ag = load_afrag(lds_wg, mt * 16, k0, 128);
            accg[mt] = wmma_bf16(ag, bfr, accg[mt]);
        }
    }

    // z = tanh(f+bf)*sigmoid(g+bg); re-layout C-frag -> B-frag via LDS
    __shared__ bf16_t zlds[NWAVES][16][DCc + 8];   // [wave][n][k], padded rows
#pragma unroll
    for (int mt = 0; mt < 4; ++mt) {
        const int mb = mt * 16 + hf * 8;           // 8 consecutive out-channels
        bf16_t zv[8];
#pragma unroll
        for (int v = 0; v < 8; ++v) {
            float fv = accf[mt][v] + bf_bias[mb + v];
            float gv = accg[mt][v] + bg_bias[mb + v];
            zv[v] = (bf16_t)gated_act(fv, gv);
        }
        *(v8bf*)&zlds[wave][lrow][mb] = *(const v8bf*)zv;  // 16B ds_store
    }
    __syncthreads();

    // residual GEMM: h_out = h_in + Wr * z + br   (K = 64, 2 chunks)
    v8f accr[4];
#pragma unroll
    for (int mt = 0; mt < 4; ++mt) accr[mt] = v8f_zero();
#pragma unroll
    for (int kc = 0; kc < 2; ++kc) {
        const int k0 = kc * 32;
        const bf16_t* zp = &zlds[wave][lrow][k0 + hf * 16];
        v8bf lo = *(const v8bf*)zp;
        v8bf hi = *(const v8bf*)(zp + 8);
        v16bf bz = __builtin_shufflevector(lo, hi, 0,1,2,3,4,5,6,7,8,9,10,11,12,13,14,15);
#pragma unroll
        for (int mt = 0; mt < 4; ++mt) {
            v16bf ar = load_afrag(lds_wr, mt * 16, k0, 64);
            accr[mt] = wmma_bf16(ar, bz, accr[mt]);
        }
    }

    const int t = t0 + lrow;
#pragma unroll
    for (int mt = 0; mt < 4; ++mt) {
        const int mb = mt * 16 + hf * 8;
        const float* hin = hb + (size_t)t * RCc + mb;
        float*       hop = ho + (size_t)t * RCc + mb;
        bf16_t hv16[8];
#pragma unroll
        for (int v = 0; v < 8; ++v) {
            float val = hin[v] + accr[mt][v] + br_bias[mb + v];
            hop[v] = val;
            hv16[v] = (bf16_t)val;
        }
        *(v8bf*)(hbo + (size_t)t * RCc + mb) = *(const v8bf*)hv16;
    }
}

// -------- input 1x1 conv: x[B][L][32] -> h0 f32 + bf16 mirror --------
__global__ void input_conv_kernel(const float* __restrict__ x,
                                  const float* __restrict__ w,   // [64][32]
                                  const float* __restrict__ bias,
                                  float* __restrict__ h0,
                                  bf16_t* __restrict__ h0b) {
    size_t idx = (size_t)blockIdx.x * blockDim.x + threadIdx.x;
    if (idx >= (size_t)Bsz * Lsz * RCc) return;
    const int oc = (int)(idx & 63);
    const size_t pos = idx >> 6;
    const float* xp = x + pos * INC;
    const float* wp = w + (size_t)oc * INC;
    float acc = bias[oc];
#pragma unroll
    for (int k = 0; k < INC; ++k) acc += wp[k] * xp[k];
    h0[idx] = acc;
    h0b[idx] = (bf16_t)acc;
}

// -------- weight f32->bf16 repack + zero skip accumulator --------
// wf/wg: [layer][out 64][K 128] with K = tap*64 + cin  (matches A-frag reads)
__global__ void prep_weights_kernel(const float* __restrict__ dil_w,
                                    const float* __restrict__ gate_w,
                                    const float* __restrict__ res_w,
                                    bf16_t* __restrict__ wf,
                                    bf16_t* __restrict__ wg,
                                    bf16_t* __restrict__ wr,
                                    float* __restrict__ skip_acc) {
    size_t idx = (size_t)blockIdx.x * blockDim.x + threadIdx.x;
    const size_t NFG = (size_t)NUM_LAYERS * 64 * 128;
    if (idx < NFG) {
        const size_t kk = idx & 127, o = (idx >> 7) & 63, i = idx >> 13;
        const size_t tap = kk >> 6, c = kk & 63;
        const size_t src = ((i * 64 + o) * 64 + c) * 2 + tap;
        wf[idx] = (bf16_t)dil_w[src];
        wg[idx] = (bf16_t)gate_w[src];
    }
    const size_t NR = (size_t)NUM_LAYERS * 64 * 64;
    if (idx < NR) wr[idx] = (bf16_t)res_w[idx];
    if (idx < (size_t)Bsz * SCc) skip_acc[idx] = 0.0f;
}

// -------- head: relu -> end1 -> mean / logvar / vol --------
__global__ void head_kernel(const float* __restrict__ skip_acc,
                            const float* __restrict__ e1w,  // [128][64]
                            const float* __restrict__ e1b,
                            const float* __restrict__ mw, const float* __restrict__ mb,
                            const float* __restrict__ lw, const float* __restrict__ lb,
                            float* __restrict__ out) {
    const int b = blockIdx.x;
    __shared__ float y1[SCc];
    __shared__ float y2[ECc];
    const int t = threadIdx.x; // 128 threads
    if (t < SCc) y1[t] = fmaxf(skip_acc[b * SCc + t], 0.0f);
    __syncthreads();
    if (t < ECc) {
        float a = e1b[t];
        for (int c = 0; c < SCc; ++c) a += e1w[t * SCc + c] * y1[c];
        y2[t] = fmaxf(a, 0.0f);
    }
    __syncthreads();
    if (t == 0) {
        float m = mb[0], l = lb[0];
        for (int c = 0; c < ECc; ++c) { m += mw[c] * y2[c]; l += lw[c] * y2[c]; }
        out[b]           = m;                    // mean_pred
        out[Bsz + b]     = l;                    // log_var_pred
        out[2 * Bsz + b] = __expf(0.5f * l);     // vol
    }
}

extern "C" void kernel_launch(void* const* d_in, const int* in_sizes, int n_in,
                              void* d_out, int out_size, void* d_ws, size_t ws_size,
                              hipStream_t stream) {
    (void)in_sizes; (void)n_in; (void)out_size; (void)ws_size;
    const float* x       = (const float*)d_in[0];
    const float* input_w = (const float*)d_in[1];
    const float* input_b = (const float*)d_in[2];
    const float* dil_w   = (const float*)d_in[3];
    const float* dil_b   = (const float*)d_in[4];
    const float* gate_w  = (const float*)d_in[5];
    const float* gate_b  = (const float*)d_in[6];
    const float* skip_w  = (const float*)d_in[7];
    const float* skip_b  = (const float*)d_in[8];
    const float* res_w   = (const float*)d_in[9];
    const float* res_b   = (const float*)d_in[10];
    const float* end1_w  = (const float*)d_in[11];
    const float* end1_b  = (const float*)d_in[12];
    const float* mean_w  = (const float*)d_in[13];
    const float* mean_b  = (const float*)d_in[14];
    const float* lv_w    = (const float*)d_in[15];
    const float* lv_b    = (const float*)d_in[16];

    // workspace carve-up (~103 MB): h f32 ping/pong + bf16 mirrors + weights
    char* ws = (char*)d_ws;
    size_t off = 0;
    auto carve = [&](size_t bytes) -> char* {
        char* p = ws + off;
        off += (bytes + 255) & ~(size_t)255;
        return p;
    };
    const size_t NH = (size_t)Bsz * Lsz * RCc;
    float*  hA       = (float*) carve(NH * sizeof(float));
    float*  hB       = (float*) carve(NH * sizeof(float));
    bf16_t* hbA      = (bf16_t*)carve(NH * sizeof(bf16_t));
    bf16_t* hbB      = (bf16_t*)carve(NH * sizeof(bf16_t));
    bf16_t* wf       = (bf16_t*)carve((size_t)NUM_LAYERS * 64 * 128 * sizeof(bf16_t));
    bf16_t* wg       = (bf16_t*)carve((size_t)NUM_LAYERS * 64 * 128 * sizeof(bf16_t));
    bf16_t* wr       = (bf16_t*)carve((size_t)NUM_LAYERS * 64 * 64 * sizeof(bf16_t));
    float*  skip_acc = (float*) carve((size_t)Bsz * SCc * sizeof(float));

    {   // weight repack + skip zeroing
        const size_t n = (size_t)NUM_LAYERS * 64 * 128;
        prep_weights_kernel<<<(unsigned)((n + 255) / 256), 256, 0, stream>>>(
            dil_w, gate_w, res_w, wf, wg, wr, skip_acc);
    }
    {   // input 1x1 conv
        input_conv_kernel<<<(unsigned)((NH + 255) / 256), 256, 0, stream>>>(
            x, input_w, input_b, hA, hbA);
    }

    float*  hin  = hA;  float*  hout  = hB;
    bf16_t* hbin = hbA; bf16_t* hbout = hbB;
    const dim3 grid(Lsz / TILE_N + 1, Bsz);   // +1 block/row for the skip tap
    for (int i = 0; i < NUM_LAYERS; ++i) {
        const int d = 1 << (i % 10);
        wavenet_layer_kernel<<<grid, 256, 0, stream>>>(
            hin, hbin, hout, hbout,
            wf + (size_t)i * 64 * 128,
            wg + (size_t)i * 64 * 128,
            wr + (size_t)i * 64 * 64,
            dil_b + i * 64, gate_b + i * 64, res_b + i * 64,
            dil_w  + (size_t)i * 64 * 64 * 2,
            gate_w + (size_t)i * 64 * 64 * 2,
            skip_w + (size_t)i * 64 * 64,
            skip_b + i * 64,
            skip_acc, d);
        float*  t1 = hin;  hin  = hout;  hout  = t1;
        bf16_t* t2 = hbin; hbin = hbout; hbout = t2;
    }

    head_kernel<<<Bsz, 128, 0, stream>>>(skip_acc, end1_w, end1_b,
                                         mean_w, mean_b, lv_w, lv_b,
                                         (float*)d_out);
}
